// FullyConnectedTensorProduct_6957847019585
// MI455X (gfx1250) — matrix operand order repaired and checked by reading, not compiled, not run on hardware
//
#include <hip/hip_runtime.h>

typedef _Float16 h16;
typedef __attribute__((ext_vector_type(16))) _Float16 v16h;
typedef __attribute__((ext_vector_type(8)))  _Float16 v8h;
typedef __attribute__((ext_vector_type(8)))  float    v8f;

// ---- real-basis Wigner 3j coefficients (e3nn convention, norm 1) ----
#define C3  0.57735026919f   // 1/sqrt(3)
#define C5  0.44721359550f   // 1/sqrt(5)
#define CA  0.31622776602f   // 1/sqrt(10)
#define CD  0.36514837167f   // 2/sqrt(30)
#define CE  0.18257418584f   // 1/sqrt(30)
#define V1  0.20701966780f   // (1/(2*sqrt(2)))/sqrt(35/12)
#define V2  0.23904572186f   // (1/sqrt(6))/sqrt(35/12)
#define V3  0.11952286093f   // (1/(2*sqrt(6)))/sqrt(35/12)

// Wt layout in d_ws (halves): lo0 at 0 (128w x 384k), lo1 at 49152 (128x512),
// lo2 at 114688 (128x512). k = p_local*128 + u, contiguous per column w.
#define WT0 0
#define WT1 49152
#define WT2 114688
#define WT_TOTAL 180224

__global__ void __launch_bounds__(256)
prep_weights(const float* __restrict__ w, h16* __restrict__ wt) {
  int idx = blockIdx.x * 256 + threadIdx.x;
  if (idx >= WT_TOTAL) return;
  int lo, K, rem; float pw;
  if (idx < WT1)      { lo = 0; K = 384; rem = idx;       pw = 0.05103103630f; }
  else if (idx < WT2) { lo = 1; K = 512; rem = idx - WT1; pw = 0.07654655446f; }
  else                { lo = 2; K = 512; rem = idx - WT2; pw = 0.09882117688f; }
  int k = rem % K;
  int pl = k >> 7, u = k & 127;
  int wcol = rem / K;
  const int pg0[3] = {0, 4, 9};
  const int pg1[4] = {1, 3, 6, 8};
  const int pg2[4] = {2, 5, 7, 10};
  int pg = (lo == 0) ? pg0[pl] : (lo == 1) ? pg1[pl] : pg2[pl];
  float v = w[pg * 16384 + u * 128 + wcol] * pw;
  wt[idx] = (h16)v;
}

// Structural nonzero mask of R per (path, i): bit k set -> R[i][k] != 0.
constexpr unsigned char RMASK[11][5] = {
  {0b00001, 0, 0, 0, 0},                           // 0  (0,0,0)
  {0b00111, 0, 0, 0, 0},                           // 1  (0,1,1)
  {0b11111, 0, 0, 0, 0},                           // 2  (0,2,2)
  {0b00001, 0b00010, 0b00100, 0, 0},               // 3  (1,0,1) diag
  {0b00001, 0b00001, 0b00001, 0, 0},               // 4  (1,1,0)
  {0b10111, 0b01110, 0b11101, 0, 0},               // 5  (1,1,2)
  {0b00111, 0b00111, 0b00111, 0, 0},               // 6  (1,2,1)
  {0b00001, 0b00010, 0b00100, 0b01000, 0b10000},   // 7  (2,0,2) diag
  {0b00101, 0b00011, 0b00111, 0b00110, 0b00101},   // 8  (2,1,1)
  {0b00001, 0b00001, 0b00001, 0b00001, 0b00001},   // 9  (2,2,0)
  {0b01111, 0b11111, 0b11111, 0b11111, 0b11110},   // 10 (2,2,2)
};

// R[i][k] = sum_j C[i,j,k] * x2_j  (x2-dependent half of the CG contraction,
// hoisted out of the per-u loop).
// q[]: [0]=l0, [1..3]=l1 (y,z,x), [4..8]=l2 (xy,yz,z2,zx,x2-y2)
__device__ __forceinline__ void path_R(int pg, const float* q, float R[5][5]) {
  switch (pg) {
    case 0: R[0][0] = q[0]; break;
    case 1: R[0][0] = C3*q[1]; R[0][1] = C3*q[2]; R[0][2] = C3*q[3]; break;
    case 2:
      for (int k = 0; k < 5; ++k) R[0][k] = C5*q[4+k];
      break;
    case 3: { float s = C3*q[0]; R[0][0]=s; R[1][1]=s; R[2][2]=s; } break;
    case 4: R[0][0] = C3*q[1]; R[1][0] = C3*q[2]; R[2][0] = C3*q[3]; break;
    case 5:
      R[0][0]= CA*q[3]; R[0][1]= CA*q[2]; R[0][2]=-CE*q[1]; R[0][4]=-CA*q[1];
      R[1][1]= CA*q[1]; R[1][2]= CD*q[2]; R[1][3]= CA*q[3];
      R[2][0]= CA*q[1]; R[2][2]=-CE*q[3]; R[2][3]= CA*q[2]; R[2][4]= CA*q[3];
      break;
    case 6:
      R[0][0]=-CE*q[6]-CA*q[8]; R[1][0]= CA*q[5]; R[2][0]= CA*q[4];
      R[0][1]= CA*q[5];         R[1][1]= CD*q[6]; R[2][1]= CA*q[7];
      R[0][2]= CA*q[4];         R[1][2]= CA*q[7]; R[2][2]=-CE*q[6]+CA*q[8];
      break;
    case 7: { float s = C5*q[0];
      R[0][0]=s; R[1][1]=s; R[2][2]=s; R[3][3]=s; R[4][4]=s; } break;
    case 8:
      R[0][0]= CA*q[3]; R[1][0]= CA*q[2]; R[2][0]=-CE*q[1]; R[4][0]=-CA*q[1];
      R[1][1]= CA*q[1]; R[2][1]= CD*q[2]; R[3][1]= CA*q[3];
      R[0][2]= CA*q[1]; R[3][2]= CA*q[2]; R[2][2]=-CE*q[3]; R[4][2]= CA*q[3];
      break;
    case 9:
      for (int m = 0; m < 5; ++m) R[m][0] = C5*q[4+m];
      break;
    case 10: {
      const float* X = q + 4;
      R[0][0]=-V2*X[2]; R[1][0]= V1*X[3];          R[2][0]=-V2*X[0];
      R[3][0]= V1*X[1];
      R[0][1]= V1*X[3]; R[1][1]= V3*X[2]-V1*X[4];  R[2][1]= V3*X[1];
      R[3][1]= V1*X[0]; R[4][1]=-V1*X[1];
      R[0][2]=-V2*X[0]; R[1][2]= V3*X[1];          R[2][2]= V2*X[2];
      R[3][2]= V3*X[3]; R[4][2]=-V2*X[4];
      R[0][3]= V1*X[1]; R[1][3]= V1*X[0];          R[2][3]= V3*X[3];
      R[3][3]= V3*X[2]+V1*X[4]; R[4][3]= V1*X[3];
      R[1][4]=-V1*X[1]; R[2][4]=-V2*X[4];          R[3][4]= V1*X[3];
      R[4][4]=-V2*X[2];
      break; }
  }
}

template <int LO> struct LoCfg;
template <> struct LoCfg<0> {
  static constexpr int NP = 3, OFF = 0;
  static constexpr int pg[3] = {0, 4, 9};
  static constexpr int l1[3] = {0, 1, 2};
};
template <> struct LoCfg<1> {
  static constexpr int NP = 4, OFF = 128;
  static constexpr int pg[4] = {1, 3, 6, 8};
  static constexpr int l1[4] = {0, 1, 1, 2};
};
template <> struct LoCfg<2> {
  static constexpr int NP = 4, OFF = 512;
  static constexpr int pg[4] = {2, 5, 7, 10};
  static constexpr int l1[4] = {0, 1, 2, 2};
};

#define ASTRIDE 136  // halves per A-row (128 + 8 pad -> 272B, conflict-free b128)

template <int LO>
__device__ __forceinline__ void lo_pass(const v8h* __restrict__ xr,  // SoA x1 rows (f16) in regs
                                        h16* __restrict__ Ash,
                                        const float* q,
                                        const h16* __restrict__ wtg,
                                        float* __restrict__ out,
                                        int n0, int tid) {
  constexpr int NK = 2 * LO + 1;
  constexpr int NP = LoCfg<LO>::NP;
  constexpr int KLO = NP * 128;
  constexpr int OFF = LoCfg<LO>::OFF;

  const int lane = tid & 31;
  const int wid  = tid >> 5;        // ct: output-column tile (w block), 0..7
  const int hw   = lane >> 4;       // half-wave
  const int l16  = lane & 15;
  const int nI   = tid >> 4;        // staging: sample index 0..15
  const int ub   = (tid & 15) * 8;  // staging: first of 8 consecutive u's

  v8f acc[NK];
#pragma unroll
  for (int m = 0; m < NK; ++m) {
    v8f z = {0.f, 0.f, 0.f, 0.f, 0.f, 0.f, 0.f, 0.f};
    acc[m] = z;
  }

#pragma unroll
  for (int p = 0; p < NP; ++p) {
    const int pg = LoCfg<LO>::pg[p];
    const int l1v = LoCfg<LO>::l1[p];
    const int d1 = 2 * l1v + 1;
    const int srow = (l1v == 0) ? 0 : (l1v == 1) ? 1 : 4;  // SoA row base

    __syncthreads();  // previous A chunk fully consumed

    // prefetch all 4 B fragments of this path (latency hidden under staging)
    v16h Bf[4];
#pragma unroll
    for (int s = 0; s < 4; ++s) {
      const h16* bp = wtg + (wid * 16 + l16) * KLO + p * 128 + s * 32 + hw * 16;
      v8h b0 = *(const v8h*)bp;
      v8h b1 = *(const v8h*)(bp + 8);
      Bf[s] = __builtin_shufflevector(b0, b1, 0, 1, 2, 3, 4, 5, 6, 7,
                                      8, 9, 10, 11, 12, 13, 14, 15);
    }

    // ---- stage A chunk: t[n,u,k] for u = ub..ub+7, packed f16 FMA ----
    float R[5][5] = {};
    path_R(pg, q, R);

    v8h t[NK];
#pragma unroll
    for (int k = 0; k < NK; ++k) {
      v8h z = {(h16)0.f, (h16)0.f, (h16)0.f, (h16)0.f,
               (h16)0.f, (h16)0.f, (h16)0.f, (h16)0.f};
      t[k] = z;
    }
#pragma unroll
    for (int i = 0; i < 5; ++i)
      if (i < d1) {
        v8h a = xr[srow + i];
#pragma unroll
        for (int k = 0; k < NK; ++k)
          if (RMASK[pg][i] & (1 << k)) t[k] += a * (h16)R[i][k];
      }
#pragma unroll
    for (int k = 0; k < NK; ++k)
      *(v8h*)(Ash + (nI * NK + k) * ASTRIDE + ub) = t[k];

    __syncthreads();

    // ---- WMMA over this K=128 chunk (4 steps of K=32) ----
#pragma unroll
    for (int s = 0; s < 4; ++s) {
#pragma unroll
      for (int m = 0; m < NK; ++m) {
        const h16* ap = Ash + (m * 16 + l16) * ASTRIDE + s * 32 + hw * 16;
        v8h a0 = *(const v8h*)ap;
        v8h a1 = *(const v8h*)(ap + 8);
        v16h A = __builtin_shufflevector(a0, a1, 0, 1, 2, 3, 4, 5, 6, 7,
                                         8, 9, 10, 11, 12, 13, 14, 15);
        acc[m] = __builtin_amdgcn_wmma_f32_16x16x32_f16(
            false, A, false, Bf[s], (short)0, acc[m], false, false);
      }
    }
  }

  // ---- store D: VGPR r, lane L -> M = r + 8*hw, N = l16 ----
  // Per-(m,r) offsets are compile-time for each half-wave; select with hw.
  float* base = out + (size_t)n0 * 1152 + OFF + (wid * 16 + l16) * NK;
#pragma unroll
  for (int m = 0; m < NK; ++m) {
#pragma unroll
    for (int r = 0; r < 8; ++r) {
      constexpr_int_trick:;
      const int row0 = m * 16 + r;       // hw = 0
      const int row1 = m * 16 + r + 8;   // hw = 1
      const int o0 = (row0 / NK) * 1152 + (row0 % NK);
      const int o1 = (row1 / NK) * 1152 + (row1 % NK);
      base[hw ? o1 : o0] = acc[m][r];
    }
  }
}

__global__ void __launch_bounds__(256)
fctp_kernel(const float* __restrict__ x1, const float* __restrict__ x2,
            const h16* __restrict__ wt, float* __restrict__ out) {
  __shared__ h16 Ash[80 * ASTRIDE];  // 21760 B: one path's t-chunk (only LDS)

  const int tid = threadIdx.x;
  const int n0 = blockIdx.x * 16;
  const int nI = tid >> 4;          // this thread's sample
  const int ub = (tid & 15) * 8;    // this thread's 8 consecutive u's

  // Load this thread's x1 slice once, SoA by spherical component, as f16:
  // rows: 0 = l0; 1..3 = l1 (i=0..2); 4..8 = l2 (i=0..4). Held in VGPRs
  // for the whole block (no cross-thread sharing exists for x1).
  const float* xg = x1 + (size_t)(n0 + nI) * 1152;
  v8h xr[9];
  {
    float4 a = *(const float4*)(xg + ub);
    float4 b = *(const float4*)(xg + ub + 4);
    v8h h;
    h[0] = (h16)a.x; h[1] = (h16)a.y; h[2] = (h16)a.z; h[3] = (h16)a.w;
    h[4] = (h16)b.x; h[5] = (h16)b.y; h[6] = (h16)b.z; h[7] = (h16)b.w;
    xr[0] = h;
  }
#pragma unroll
  for (int i = 0; i < 3; ++i) {
    v8h h;
#pragma unroll
    for (int j = 0; j < 8; ++j) h[j] = (h16)xg[128 + (ub + j) * 3 + i];
    xr[1 + i] = h;
  }
#pragma unroll
  for (int i = 0; i < 5; ++i) {
    v8h h;
#pragma unroll
    for (int j = 0; j < 8; ++j) h[j] = (h16)xg[512 + (ub + j) * 5 + i];
    xr[4 + i] = h;
  }

  float q[9];
#pragma unroll
  for (int j = 0; j < 9; ++j) q[j] = x2[(n0 + nI) * 9 + j];

  lo_pass<0>(xr, Ash, q, wt + WT0, out, n0, tid);
  lo_pass<1>(xr, Ash, q, wt + WT1, out, n0, tid);
  lo_pass<2>(xr, Ash, q, wt + WT2, out, n0, tid);
}

extern "C" void kernel_launch(void* const* d_in, const int* in_sizes, int n_in,
                              void* d_out, int out_size, void* d_ws, size_t ws_size,
                              hipStream_t stream) {
  const float* x1 = (const float*)d_in[0];
  const float* x2 = (const float*)d_in[1];
  const float* w  = (const float*)d_in[2];
  float* out = (float*)d_out;
  h16* wt = (h16*)d_ws;  // needs 360448 bytes of scratch

  int n = in_sizes[0] / 1152;   // 20000
  int nblk = n / 16;            // 1250 (N is a multiple of 16 here)

  prep_weights<<<(WT_TOTAL + 255) / 256, 256, 0, stream>>>(w, wt);
  fctp_kernel<<<nblk, 256, 0, stream>>>(x1, x2, wt, out);
}
